// GConv_29729763623145
// MI455X (gfx1250) — compile-verified, matrix operand order checked by reading, not run on hardware
//
#include <hip/hip_runtime.h>
#include <hip/hip_bf16.h>
#include <stdint.h>

// CDNA5 / gfx1250: wave32, WMMA 16x16x32 bf16 with f32 accumulation.
typedef __attribute__((ext_vector_type(16))) __bf16 v16bf;
typedef __attribute__((ext_vector_type(8)))  float  v8f;
typedef unsigned int __attribute__((ext_vector_type(4))) uint32x4;
typedef int __attribute__((ext_vector_type(4)))  int32x4;
typedef int __attribute__((ext_vector_type(8)))  int32x8;

#define N_   1024
#define F_   64
#define K_   16        // MIN_NN == RANK == 16
#define NT_  4096      // B*N nodes == edge tiles of 16
#define E_   65536     // NT_ * K_

#if __has_builtin(__builtin_amdgcn_tensor_load_to_lds) && __has_builtin(__builtin_amdgcn_s_wait_tensorcnt)
#define CDNA5_HAS_TDM 1
#else
#define CDNA5_HAS_TDM 0
#endif

__device__ __forceinline__ v8f wmma_bf16(v16bf a, v16bf b, v8f c) {
  return __builtin_amdgcn_wmma_f32_16x16x32_bf16(false, a, false, b, (short)0, c, false, false);
}

// K index inside a 16x32 bf16 A (or mirrored B) fragment:
// VGPR v<4: K = 2v+8h(+p) ; v>=4: K = 16+2(v-4)+8h(+p)
__device__ __host__ __forceinline__ int kloc(int v, int h, int p) {
  return (v < 4 ? 2 * v : 16 + 2 * (v - 4)) + 8 * h + p;
}

// Stage 128 KB of swizzled weights into LDS.
// Preferred path: single Tensor-Data-Mover descriptor (2D, 16384 x 8B, tile==tensor),
// issued by wave 0 (TDM ignores EXEC; one issue per wave), completion via TENSORcnt.
__device__ __forceinline__ void stage_weights(const __bf16* __restrict__ gsrc,
                                              __bf16* lds_dst, int tid) {
#if CDNA5_HAS_TDM
  if (tid < 32) {
    unsigned ldsa = (unsigned)(unsigned long long)
        (__attribute__((address_space(3))) char*)(void*)lds_dst;
    unsigned long long ga = (unsigned long long)gsrc;
    uint32x4 g0 = { 1u,                                    // count=1, user mode
                    ldsa,                                  // lds_addr (bytes)
                    (unsigned)(ga & 0xffffffffu),          // global_addr[31:0]
                    (unsigned)((ga >> 32) & 0x01ffffffu) | (2u << 30) }; // [56:32] | type=2
    int32x8 g1;
    g1[0] = 3 << 16;              // workgroup_mask=0, data_size=3 (8 bytes)
    g1[1] = (int)(16384u << 16);  // tensor_dim0[15:0]=0x4000 at bits[31:16]
    g1[2] = 1 << 16;              // tensor_dim0[31:16]=0 ; tensor_dim1[15:0]=1
    g1[3] = (int)(16384u << 16);  // tensor_dim1[31:16]=0 ; tile_dim0=16384
    g1[4] = 1;                    // tile_dim1=1, tile_dim2=0
    g1[5] = 16384;                // tensor_dim0_stride[31:0]
    g1[6] = 0;                    // stride0 hi, stride1 lo
    g1[7] = 0;                    // stride1 hi
    int32x4 zz = {0, 0, 0, 0};
#if __clang_major__ >= 23
    int32x8 z8 = {0, 0, 0, 0, 0, 0, 0, 0};
    __builtin_amdgcn_tensor_load_to_lds(g0, g1, zz, zz, z8, 0);
#else
    __builtin_amdgcn_tensor_load_to_lds(g0, g1, zz, zz, 0);
#endif
    __builtin_amdgcn_s_wait_tensorcnt(0);
  }
#else
  const uint4* src = (const uint4*)gsrc;
  uint4* dst = (uint4*)lds_dst;
  for (int i = tid; i < 8192; i += 256) dst[i] = src[i];
#endif
}

// ---------------------------------------------------------------- kernel 1
// per (b,n): 17 smallest of D row -> self index + 16 neighbor indices
__global__ void topk_kernel(const float* __restrict__ D, int* __restrict__ selfidx,
                            int* __restrict__ nbr) {
  extern __shared__ char s_lds[];
  float* vals = (float*)s_lds;          // 1024
  float* rv   = vals + N_;              // 256
  int*   ri   = (int*)(rv + 256);       // 256
  const int bn = blockIdx.x;
  const int t  = threadIdx.x;
  const float* row = D + (size_t)bn * N_;
  for (int j = t; j < N_; j += 256) vals[j] = row[j];
  __syncthreads();
  for (int it = 0; it < K_ + 1; ++it) {
    float bv = 3.0e38f; int bj = 0x7fffffff;
    for (int j = t; j < N_; j += 256) {
      float v = vals[j];
      if (v < bv) { bv = v; bj = j; }        // ascending scan => lowest idx on tie
    }
    rv[t] = bv; ri[t] = bj;
    __syncthreads();
    for (int s = 128; s > 0; s >>= 1) {
      if (t < s) {
        float ov = rv[t + s]; int oj = ri[t + s];
        if (ov < rv[t] || (ov == rv[t] && oj < ri[t])) { rv[t] = ov; ri[t] = oj; }
      }
      __syncthreads();
    }
    if (t == 0) {
      int idx = ri[0];
      if (it == 0) selfidx[bn] = idx; else nbr[bn * K_ + (it - 1)] = idx;
      vals[idx] = 3.0e38f;
    }
    __syncthreads();
  }
}

// ---------------------------------------------------------------- kernel 2
// Build bf16 weight tables in B-fragment-swizzled layout.
// Wb[i][q] = W[(i + shift(q/256)) % 64][q % 256], shift = {0,32,48,0}
// Fragment storage: frag*(32*16) + lane*16 + j   (j indexes v16bf elements)
__global__ void swz_kernel(const float* __restrict__ W0, const float* __restrict__ W1,
                           const float* __restrict__ W2, const float* __restrict__ Wl,
                           __bf16* __restrict__ w0s, __bf16* __restrict__ wls,
                           __bf16* __restrict__ wb1, __bf16* __restrict__ wb2) {
  int tid = blockIdx.x * blockDim.x + threadIdx.x;
  int table, t = tid;
  if (t < 4096)        { table = 0; }                 // W0  : 4 ntile * 2 kf
  else if (t < 5120)   { table = 1; t -= 4096; }      // Wl  : 1 ntile * 2 kf
  else if (t < 70656)  { table = 2; t -= 5120; }      // Wb1 : 64 ntile * 2 kf
  else if (t < 136192) { table = 3; t -= 70656; }     // Wb2
  else return;
  int fid = t >> 9, rem = t & 511, l = rem >> 4, j = rem & 15;
  int nt = fid >> 1, kf = fid & 1;
  int n = nt * 16 + (l & 15), h = l >> 4;
  int k = kf * 32 + kloc(j >> 1, h, j & 1);
  if (table == 0) { w0s[t] = (__bf16)W0[k * 64 + n]; return; }
  if (table == 1) { wls[t] = (__bf16)Wl[k * 16 + n]; return; }
  const float* W = (table == 2) ? W1 : W2;
  int roll = n >> 8;
  int s = (roll == 1) ? 32 : (roll == 2) ? 48 : 0;
  float val = W[((k + s) & 63) * 256 + (n & 255)];
  ((table == 2) ? wb1 : wb2)[t] = (__bf16)val;
}

// ---------------------------------------------------------------- kernel 3
// Per node (== 16-edge tile, one wave): labels, x1, w=exp(-|labels|^2/10),
// lab = leakyrelu(labels@W0+b0) via WMMA, emit lab as bf16 A-fragments.
__global__ void edge_kernel(const float* __restrict__ h, const int* __restrict__ selfidx,
                            const int* __restrict__ nbr, const float* __restrict__ b0,
                            const __bf16* __restrict__ w0s, float* __restrict__ x1ws,
                            float* __restrict__ wws, __bf16* __restrict__ labA) {
  extern __shared__ char s_lds[];
  const int lane = threadIdx.x & 31;
  const int wid  = threadIdx.x >> 5;
  const int bn   = blockIdx.x * 8 + wid;
  float* labt = (float*)s_lds + wid * (16 * F_);   // [16 e][64 f] per wave
  const int b = bn >> 10;
  const int si = selfidx[bn];
  const float2 x2 = *(const float2*)(h + ((size_t)(b * N_ + si)) * F_ + 2 * lane);
  for (int k = 0; k < K_; ++k) {
    const int e = bn * K_ + k;
    const int r = nbr[e];
    const float2 x1v = *(const float2*)(h + ((size_t)(b * N_ + r)) * F_ + 2 * lane);
    *(float2*)(x1ws + (size_t)e * F_ + 2 * lane) = x1v;
    float d0 = x1v.x - x2.x, d1 = x1v.y - x2.y;
    labt[k * F_ + 2 * lane]     = d0;
    labt[k * F_ + 2 * lane + 1] = d1;
    float ss = d0 * d0 + d1 * d1;
    ss += __shfl_xor(ss, 1, 32);  ss += __shfl_xor(ss, 2, 32);
    ss += __shfl_xor(ss, 4, 32);  ss += __shfl_xor(ss, 8, 32);
    ss += __shfl_xor(ss, 16, 32);
    if (lane == 0) wws[e] = __expf(-ss * 0.1f);
  }
  const int m = lane & 15, hh = lane >> 4;
  v16bf a0, a1;
#pragma unroll
  for (int j = 0; j < 16; ++j) {
    int kl = kloc(j >> 1, hh, j & 1);
    a0[j] = (__bf16)labt[m * F_ + kl];
    a1[j] = (__bf16)labt[m * F_ + 32 + kl];
  }
#pragma unroll
  for (int nt = 0; nt < 4; ++nt) {
    v16bf bb0 = *(const v16bf*)(w0s + ((nt * 2 + 0) * 32 + lane) * 16);
    v16bf bb1 = *(const v16bf*)(w0s + ((nt * 2 + 1) * 32 + lane) * 16);
    float bias = b0[nt * 16 + m];
    v8f c;
#pragma unroll
    for (int v = 0; v < 8; ++v) c[v] = bias;
    c = wmma_bf16(a0, bb0, c);
    c = wmma_bf16(a1, bb1, c);
#pragma unroll
    for (int v = 0; v < 8; ++v) {
      float y = c[v];
      y = y > 0.f ? y : 0.02f * y;
      labt[(v + 8 * hh) * F_ + nt * 16 + m] = y;    // labt now holds lab
    }
  }
  v16bf l0, l1;
#pragma unroll
  for (int j = 0; j < 16; ++j) {
    int kl = kloc(j >> 1, hh, j & 1);
    l0[j] = (__bf16)labt[m * F_ + kl];
    l1[j] = (__bf16)labt[m * F_ + 32 + kl];
  }
  *(v16bf*)(labA + ((size_t)(bn * 2 + 0) * 32 + lane) * 16) = l0;
  *(v16bf*)(labA + ((size_t)(bn * 2 + 1) * 32 + lane) * 16) = l1;
}

// ---------------------------------------------------------------- kernel 4
// x[e,r] = (sum_f (lab@Wb1 + b1)[e,r,f] * x1[e,f]) * (lab@Wl + bl)[e,r]
__global__ void stage1_kernel(const __bf16* __restrict__ labA, const __bf16* __restrict__ wb1s,
                              const __bf16* __restrict__ wls, const float* __restrict__ b1g,
                              const float* __restrict__ blg, const float* __restrict__ x1ws,
                              float* __restrict__ xws) {
  extern __shared__ char s_lds[];
  __bf16* wb  = (__bf16*)s_lds;                                    // 128 KB staged Wb1
  float*  thl = (float*)(s_lds + 131072) + (threadIdx.x >> 5) * 256;
  stage_weights(wb1s, wb, threadIdx.x);
  __syncthreads();
  const int lane = threadIdx.x & 31;
  const int m = lane & 15, hh = lane >> 4;
  const int tile = blockIdx.x * 8 + (threadIdx.x >> 5);
  v16bf wl0 = *(const v16bf*)(wls + (0 * 32 + lane) * 16);
  v16bf wl1 = *(const v16bf*)(wls + (1 * 32 + lane) * 16);
  v16bf a0 = *(const v16bf*)(labA + ((size_t)(tile * 2 + 0) * 32 + lane) * 16);
  v16bf a1 = *(const v16bf*)(labA + ((size_t)(tile * 2 + 1) * 32 + lane) * 16);
  float x1r[32];
#pragma unroll
  for (int v = 0; v < 8; ++v)
#pragma unroll
    for (int ft = 0; ft < 4; ++ft)
      x1r[v * 4 + ft] = x1ws[((size_t)tile * 16 + v + 8 * hh) * F_ + ft * 16 + m];
  // thetal fragment: [e, r]
  float blv = blg[m];
  v8f ct;
#pragma unroll
  for (int v = 0; v < 8; ++v) ct[v] = blv;
  ct = wmma_bf16(a0, wl0, ct);
  ct = wmma_bf16(a1, wl1, ct);
#pragma unroll
  for (int v = 0; v < 8; ++v) thl[(v + 8 * hh) * 16 + m] = ct[v];
  v8f xacc = {};
#pragma unroll 4
  for (int r = 0; r < 16; ++r) {
    v8f acc = {};
#pragma unroll
    for (int ft = 0; ft < 4; ++ft) {
      const int t = r * 4 + ft;
      v16bf bb0 = *(const v16bf*)(wb + ((t * 2 + 0) * 32 + lane) * 16);
      v16bf bb1 = *(const v16bf*)(wb + ((t * 2 + 1) * 32 + lane) * 16);
      v8f cc = {};
      cc = wmma_bf16(a0, bb0, cc);
      cc = wmma_bf16(a1, bb1, cc);
      float b1v = b1g[r * 64 + ft * 16 + m];
#pragma unroll
      for (int v = 0; v < 8; ++v) acc[v] += (cc[v] + b1v) * x1r[v * 4 + ft];
    }
#pragma unroll
    for (int v = 0; v < 8; ++v) {
      float s = acc[v];
      s += __shfl_xor(s, 1, 32);  s += __shfl_xor(s, 2, 32);
      s += __shfl_xor(s, 4, 32);  s += __shfl_xor(s, 8, 32);
      float xf = s * thl[(v + 8 * hh) * 16 + r];
      xacc[v] = (m == r) ? xf : xacc[v];
    }
  }
#pragma unroll
  for (int v = 0; v < 8; ++v)
    xws[((size_t)tile * 16 + v + 8 * hh) * 16 + m] = xacc[v];
}

// ---------------------------------------------------------------- kernel 5
// out[bn,o] = sum_k w[e] * sum_r (lab@Wb2 + b2)[e,r,o] * x[e,r]
__global__ void stage2_kernel(const __bf16* __restrict__ labA, const __bf16* __restrict__ wb2s,
                              const float* __restrict__ b2g, const float* __restrict__ xws,
                              const float* __restrict__ wws, float* __restrict__ out) {
  extern __shared__ char s_lds[];
  __bf16* wb  = (__bf16*)s_lds;                                    // 128 KB staged Wb2
  float*  xsh = (float*)(s_lds + 131072) + (threadIdx.x >> 5) * 256;
  stage_weights(wb2s, wb, threadIdx.x);
  __syncthreads();
  const int lane = threadIdx.x & 31;
  const int m = lane & 15, hh = lane >> 4;
  const int tile = blockIdx.x * 8 + (threadIdx.x >> 5);
  v16bf a0 = *(const v16bf*)(labA + ((size_t)(tile * 2 + 0) * 32 + lane) * 16);
  v16bf a1 = *(const v16bf*)(labA + ((size_t)(tile * 2 + 1) * 32 + lane) * 16);
  float wreg[8];
#pragma unroll
  for (int v = 0; v < 8; ++v) {
    xsh[(v + 8 * hh) * 16 + m] = xws[((size_t)tile * 16 + v + 8 * hh) * 16 + m];
    wreg[v] = wws[tile * 16 + v + 8 * hh];
  }
  v8f o4[4];
#pragma unroll
  for (int ot = 0; ot < 4; ++ot) o4[ot] = (v8f){};
#pragma unroll 4
  for (int r = 0; r < 16; ++r) {
    float xr[8];
#pragma unroll
    for (int v = 0; v < 8; ++v) xr[v] = xsh[(v + 8 * hh) * 16 + r];
#pragma unroll
    for (int ot = 0; ot < 4; ++ot) {
      const int t = r * 4 + ot;
      v16bf bb0 = *(const v16bf*)(wb + ((t * 2 + 0) * 32 + lane) * 16);
      v16bf bb1 = *(const v16bf*)(wb + ((t * 2 + 1) * 32 + lane) * 16);
      v8f cc = {};
      cc = wmma_bf16(a0, bb0, cc);
      cc = wmma_bf16(a1, bb1, cc);
      float b2v = b2g[r * 64 + ot * 16 + m];
#pragma unroll
      for (int v = 0; v < 8; ++v) o4[ot][v] += (cc[v] + b2v) * xr[v];
    }
  }
#pragma unroll
  for (int ot = 0; ot < 4; ++ot) {
    float s = 0.f;
#pragma unroll
    for (int v = 0; v < 8; ++v) s += o4[ot][v] * wreg[v];
    s += __shfl_xor(s, 16, 32);                 // combine e=0..7 and e=8..15 halves
    if (lane < 16) out[(size_t)tile * 64 + ot * 16 + lane] = s;
  }
}

// ---------------------------------------------------------------- launch
extern "C" void kernel_launch(void* const* d_in, const int* in_sizes, int n_in,
                              void* d_out, int out_size, void* d_ws, size_t ws_size,
                              hipStream_t stream) {
  const float* h  = (const float*)d_in[0];
  const float* D  = (const float*)d_in[1];
  const float* W0 = (const float*)d_in[2];
  const float* b0 = (const float*)d_in[3];
  const float* W1 = (const float*)d_in[4];
  const float* b1 = (const float*)d_in[5];
  const float* W2 = (const float*)d_in[6];
  const float* b2 = (const float*)d_in[7];
  const float* Wl = (const float*)d_in[8];
  const float* bl = (const float*)d_in[9];
  float* out = (float*)d_out;

  uint8_t* ws = (uint8_t*)d_ws;
  size_t off = 0;
  auto alloc = [&](size_t bytes) -> void* {
    void* p = ws + off;
    off = (off + bytes + 255) & ~(size_t)255;
    return p;
  };
  int*    selfidx = (int*)   alloc((size_t)NT_ * 4);
  int*    nbr     = (int*)   alloc((size_t)E_ * 4);
  float*  wws     = (float*) alloc((size_t)E_ * 4);
  float*  x1ws    = (float*) alloc((size_t)E_ * F_ * 4);      // 16 MB
  float*  xws     = (float*) alloc((size_t)E_ * K_ * 4);      // 4 MB
  __bf16* labA    = (__bf16*)alloc((size_t)NT_ * 2 * 512 * 2); // 8 MB
  __bf16* w0s     = (__bf16*)alloc(4096 * 2);
  __bf16* wls     = (__bf16*)alloc(1024 * 2);
  __bf16* wb1s    = (__bf16*)alloc(65536 * 2);
  __bf16* wb2s    = (__bf16*)alloc(65536 * 2);
  // total ~30 MB of workspace

  const int bigLds = 131072 + 8 * 1024;
  hipFuncSetAttribute((const void*)stage1_kernel, hipFuncAttributeMaxDynamicSharedMemorySize, bigLds);
  hipFuncSetAttribute((const void*)stage2_kernel, hipFuncAttributeMaxDynamicSharedMemorySize, bigLds);

  topk_kernel<<<NT_, 256, (N_ + 256) * 4 + 256 * 4, stream>>>(D, selfidx, nbr);
  swz_kernel<<<(136192 + 255) / 256, 256, 0, stream>>>(W0, W1, W2, Wl, w0s, wls, wb1s, wb2s);
  edge_kernel<<<NT_ / 8, 256, 8 * 16 * F_ * 4, stream>>>(h, selfidx, nbr, b0, w0s, x1ws, wws, labA);
  stage1_kernel<<<NT_ / 8, 256, bigLds, stream>>>(labA, wb1s, wls, b1, bl, x1ws, xws);
  stage2_kernel<<<NT_ / 8, 256, bigLds, stream>>>(labA, wb2s, b2, xws, wws, out);
}
